// Angleproto_34522947125433
// MI455X (gfx1250) — matrix operand order repaired and checked by reading, not compiled, not run on hardware
//
#include <hip/hip_runtime.h>

typedef __attribute__((ext_vector_type(16))) __bf16 v16bf;
typedef __attribute__((ext_vector_type(8)))  __bf16 v8bf;
typedef __attribute__((ext_vector_type(8)))  float  v8f;

#define NROWS 8192
#define KDIM  512
#define CS    8                 // column chunks per row-block
#define CHUNK (NROWS / CS)      // 1024 columns per chunk
#define NT    4                 // 16-wide column tiles held in registers (N=64)

__device__ __forceinline__ unsigned short f2bf_rne(float f) {
    unsigned int u = __float_as_uint(f);
    u += 0x7FFFu + ((u >> 16) & 1u);
    return (unsigned short)(u >> 16);
}
__device__ __forceinline__ float bf2f(unsigned short h) {
    return __uint_as_float((unsigned int)h << 16);
}

// ---------------------------------------------------------------------------
// Kernel 1: per-row L2 normalize, split even/odd rows into anchor/positive,
// decompose each normalized value into bf16 hi + bf16 lo (error term).
// One wave32 per source row.
// ---------------------------------------------------------------------------
__global__ void __launch_bounds__(256)
prep_kernel(const float* __restrict__ x,
            unsigned short* __restrict__ anc_hi, unsigned short* __restrict__ anc_lo,
            unsigned short* __restrict__ pos_hi, unsigned short* __restrict__ pos_lo) {
    const int wave = (blockIdx.x * blockDim.x + threadIdx.x) >> 5;   // 0..16383
    const int lane = threadIdx.x & 31;
    const float* row = x + (size_t)wave * KDIM;

    float ss = 0.f;
    for (int k = lane; k < KDIM; k += 32) { float v = row[k]; ss += v * v; }
    #pragma unroll
    for (int off = 16; off >= 1; off >>= 1) ss += __shfl_xor(ss, off, 32);

    float invn = rsqrtf(fmaxf(ss, 1e-24f));
    invn = invn * (1.5f - 0.5f * ss * invn * invn);   // one NR refinement

    unsigned short* dhi; unsigned short* dlo;
    const int drow = wave >> 1;
    if (wave & 1) { dhi = pos_hi; dlo = pos_lo; }
    else          { dhi = anc_hi; dlo = anc_lo; }
    dhi += (size_t)drow * KDIM;
    dlo += (size_t)drow * KDIM;

    for (int k = lane; k < KDIM; k += 32) {
        float y = row[k] * invn;
        unsigned short h = f2bf_rne(y);
        dhi[k] = h;
        dlo[k] = f2bf_rne(y - bf2f(h));
    }
}

// ---------------------------------------------------------------------------
// Kernel 2: WMMA GEMM (positive x anchor^T) with compensated split-bf16
// (hi*hi + hi*lo + lo*hi ~ fp32), fused online softmax over columns.
// One wave32 per workgroup; each wave owns a 16(M) x 64(N) register block
// (4 accumulator tiles) so the LDS A-operand loads are amortized 4x and the
// 4 accumulation chains are independent. Grid = (rowBlocks=512, colChunks=8).
// ---------------------------------------------------------------------------
__global__ void __launch_bounds__(32)
gemm_softmax_kernel(const unsigned short* __restrict__ pos_hi,
                    const unsigned short* __restrict__ pos_lo,
                    const unsigned short* __restrict__ anc_hi,
                    const unsigned short* __restrict__ anc_lo,
                    const float* __restrict__ wp, const float* __restrict__ bp,
                    float* __restrict__ diag,
                    float* __restrict__ partM, float* __restrict__ partS) {
    // [hi/lo][row][k], +8 halves row padding to spread LDS banks
    __shared__ unsigned short ldsA[2][16][520];

    const int m0   = blockIdx.x * 16;
    const int cy   = blockIdx.y;
    const int lane = threadIdx.x;
    const float w  = wp[0];
    const float b  = bp[0];

    // Stage the 16 positive rows (hi+lo) into LDS, 16B chunks.
    for (int i = lane; i < 16 * 64; i += 32) {
        const int r = i >> 6;          // 0..15
        const int c = (i & 63) * 8;    // halves, multiple of 8 -> 16B aligned
        *(v8bf*)&ldsA[0][r][c] = *(const v8bf*)&pos_hi[(size_t)(m0 + r) * KDIM + c];
        *(v8bf*)&ldsA[1][r][c] = *(const v8bf*)&pos_lo[(size_t)(m0 + r) * KDIM + c];
    }
    __syncthreads();

    const int half = lane >> 4;   // which 16-lane half
    const int l15  = lane & 15;

    float runM[8], runS[8];
    #pragma unroll
    for (int r = 0; r < 8; ++r) { runM[r] = -3.402823466e38f; runS[r] = 0.f; }

    for (int n0 = cy * CHUNK; n0 < (cy + 1) * CHUNK; n0 += 16 * NT) {
        v8f acc[NT];
        #pragma unroll
        for (int t = 0; t < NT; ++t) acc[t] = (v8f){};

        // B operand bases: lane l15 -> anchor row (n0 + t*16 + l15);
        // halves split K 0..15 / 16..31 of each 32-wide k-tile.
        const unsigned short* bhiBase[NT];
        const unsigned short* bloBase[NT];
        #pragma unroll
        for (int t = 0; t < NT; ++t) {
            bhiBase[t] = anc_hi + (size_t)(n0 + t * 16 + l15) * KDIM + half * 16;
            bloBase[t] = anc_lo + (size_t)(n0 + t * 16 + l15) * KDIM + half * 16;
        }

        #pragma unroll 2
        for (int kt = 0; kt < 16; ++kt) {
            const int k0 = kt * 32;
            // A operand (16x32, this code's own contiguous k order):
            // lanes 0-15: K0-15 ; lanes 16-31: K16-31 (contiguous, matches B)
            v16bf ahi, alo;
            {
                const v8bf a0 = *(const v8bf*)&ldsA[0][l15][k0 + half * 16];
                const v8bf a1 = *(const v8bf*)&ldsA[0][l15][k0 + half * 16 + 8];
                const v8bf e0 = *(const v8bf*)&ldsA[1][l15][k0 + half * 16];
                const v8bf e1 = *(const v8bf*)&ldsA[1][l15][k0 + half * 16 + 8];
                #pragma unroll
                for (int e = 0; e < 8; ++e) {
                    ahi[e] = a0[e]; ahi[8 + e] = a1[e];
                    alo[e] = e0[e]; alo[8 + e] = e1[e];
                }
            }
            #pragma unroll
            for (int t = 0; t < NT; ++t) {
                const v16bf bhi = *(const v16bf*)(bhiBase[t] + k0);
                const v16bf blo = *(const v16bf*)(bloBase[t] + k0);
                acc[t] = __builtin_amdgcn_wmma_f32_16x16x32_bf16(false, ahi, false, bhi,
                                                                 (short)0, acc[t], false, false);
                acc[t] = __builtin_amdgcn_wmma_f32_16x16x32_bf16(false, ahi, false, blo,
                                                                 (short)0, acc[t], false, false);
                acc[t] = __builtin_amdgcn_wmma_f32_16x16x32_bf16(false, alo, false, bhi,
                                                                 (short)0, acc[t], false, false);
            }
        }

        // Online softmax update on the C layout:
        // VGPR r holds row m0+r (lanes 0-15) / m0+8+r (lanes 16-31),
        // col = n0 + t*16 + l15.
        #pragma unroll
        for (int t = 0; t < NT; ++t) {
            #pragma unroll
            for (int r = 0; r < 8; ++r) {
                const float v    = acc[t][r] * w + b;
                const int   rowg = m0 + r + half * 8;
                const int   colg = n0 + t * 16 + l15;
                if (rowg == colg) diag[rowg] = v;   // exactly one lane, once total

                float bm = v;
                #pragma unroll
                for (int off = 1; off < 16; off <<= 1) bm = fmaxf(bm, __shfl_xor(bm, off, 32));
                const float nM = fmaxf(runM[r], bm);
                float s = __expf(v - nM);
                #pragma unroll
                for (int off = 1; off < 16; off <<= 1) s += __shfl_xor(s, off, 32);
                runS[r] = runS[r] * __expf(runM[r] - nM) + s;
                runM[r] = nM;
            }
        }
    }

    if (l15 == 0) {
        const int base = m0 + half * 8;
        #pragma unroll
        for (int r = 0; r < 8; ++r) {
            partM[(size_t)(base + r) * CS + cy] = runM[r];
            partS[(size_t)(base + r) * CS + cy] = runS[r];
        }
    }
}

// ---------------------------------------------------------------------------
// Kernel 3: combine per-chunk softmax partials; per-row NLL term.
// ---------------------------------------------------------------------------
__global__ void __launch_bounds__(256)
rowterm_kernel(const float* __restrict__ partM, const float* __restrict__ partS,
               const float* __restrict__ diag, float* __restrict__ rowterm) {
    const int i = blockIdx.x * blockDim.x + threadIdx.x;
    if (i >= NROWS) return;
    float M = -3.402823466e38f;
    #pragma unroll
    for (int c = 0; c < CS; ++c) M = fmaxf(M, partM[(size_t)i * CS + c]);
    float S = 0.f;
    #pragma unroll
    for (int c = 0; c < CS; ++c)
        S += partS[(size_t)i * CS + c] * __expf(partM[(size_t)i * CS + c] - M);
    rowterm[i] = -(diag[i] - M - logf(S));
}

// ---------------------------------------------------------------------------
// Kernel 4: mean over rows -> scalar loss.
// ---------------------------------------------------------------------------
__global__ void __launch_bounds__(256)
reduce_kernel(const float* __restrict__ rowterm, float* __restrict__ out) {
    __shared__ float sm[256];
    float s = 0.f;
    for (int i = threadIdx.x; i < NROWS; i += 256) s += rowterm[i];
    sm[threadIdx.x] = s;
    __syncthreads();
    for (int off = 128; off >= 1; off >>= 1) {
        if ((int)threadIdx.x < off) sm[threadIdx.x] += sm[threadIdx.x + off];
        __syncthreads();
    }
    if (threadIdx.x == 0) out[0] = sm[0] / (float)NROWS;
}

// ---------------------------------------------------------------------------
extern "C" void kernel_launch(void* const* d_in, const int* in_sizes, int n_in,
                              void* d_out, int out_size, void* d_ws, size_t ws_size,
                              hipStream_t stream) {
    const float* x  = (const float*)d_in[0];
    const float* wp = (const float*)d_in[1];
    const float* bp = (const float*)d_in[2];

    char* p = (char*)d_ws;
    unsigned short* anc_hi = (unsigned short*)p; p += (size_t)NROWS * KDIM * 2;
    unsigned short* anc_lo = (unsigned short*)p; p += (size_t)NROWS * KDIM * 2;
    unsigned short* pos_hi = (unsigned short*)p; p += (size_t)NROWS * KDIM * 2;
    unsigned short* pos_lo = (unsigned short*)p; p += (size_t)NROWS * KDIM * 2;
    float* diag    = (float*)p; p += (size_t)NROWS * 4;
    float* partM   = (float*)p; p += (size_t)NROWS * CS * 4;
    float* partS   = (float*)p; p += (size_t)NROWS * CS * 4;
    float* rowterm = (float*)p; p += (size_t)NROWS * 4;

    prep_kernel<<<dim3(2 * NROWS / 8), dim3(256), 0, stream>>>(
        x, anc_hi, anc_lo, pos_hi, pos_lo);

    gemm_softmax_kernel<<<dim3(NROWS / 16, CS), dim3(32), 0, stream>>>(
        pos_hi, pos_lo, anc_hi, anc_lo, wp, bp, diag, partM, partS);

    rowterm_kernel<<<dim3(NROWS / 256), dim3(256), 0, stream>>>(
        partM, partS, diag, rowterm);

    reduce_kernel<<<dim3(1), dim3(256), 0, stream>>>(rowterm, d_out ? (float*)d_out : (float*)d_ws);
}
